// BRepNetEncoder_56049323213417
// MI455X (gfx1250) — matrix operand order repaired and checked by reading, not compiled, not run on hardware
//
#include <hip/hip_runtime.h>
#include <hip/hip_bf16.h>
#include <stdint.h>

typedef _Float16 v16h __attribute__((ext_vector_type(16)));
typedef _Float16 v8h  __attribute__((ext_vector_type(8)));
typedef _Float16 v4h  __attribute__((ext_vector_type(4)));
typedef float    v8f  __attribute__((ext_vector_type(8)));

#define NCO      500000
#define NFACE    50000
#define DIM      128
#define INDIM    12
#define NLAYERS  6
#define LN_EPS   1e-5f

#define NTILES   (NCO/16)          // 31250, exact
#define MT       4                 // row-tiles (16 rows each) per wave
#define NGROUPS  ((NTILES + MT - 1) / MT)   // 7813
#define FRAG_HALF 512              // 32 lanes * 16 halves per (t,c) fragment
#define LAYER_FRAG_HALVES (4*8*4*FRAG_HALF)           // 65536 per layer
#define ALL_LAYER_HALVES  (NLAYERS*LAYER_FRAG_HALVES) // 393216
#define OUT_FRAG_HALVES   (8*4*FRAG_HALF)             // 16384

// ---------------------------------------------------------------------------
// K0: pack fp32 weights into WMMA B-fragment order (f16).
// B 32x16 layout (16x16x32 f16 WMMA): lane half h=lane/16, col n=lane%16;
// k = 8h+e for e<8, k = 16+8h+(e-8) for e>=8. k-chunk c adds c*32.
// halves index = (((l*4+s)*8+t)*4+c)*512 + lane*16 + e
// ---------------------------------------------------------------------------
__global__ void pack_kernel(const float* __restrict__ Ws,
                            const float* __restrict__ out_w,
                            _Float16* __restrict__ wfragL,
                            _Float16* __restrict__ wfragO) {
  int tid = blockIdx.x * blockDim.x + threadIdx.x;
  if (tid < ALL_LAYER_HALVES) {
    int e    = tid & 15;
    int lane = (tid >> 4) & 31;
    int c    = (tid >> 9) & 3;
    int t    = (tid >> 11) & 7;
    int s    = (tid >> 14) & 3;
    int l    = tid >> 16;
    int k = c * 32 + ((lane < 16) ? 0 : 8) + e + ((e < 8) ? 0 : 8);
    int n = t * 16 + (lane & 15);
    wfragL[tid] = (_Float16)Ws[(((l * 4 + s) * 128) + k) * 128 + n];
  } else if (tid < ALL_LAYER_HALVES + OUT_FRAG_HALVES) {
    int t2   = tid - ALL_LAYER_HALVES;
    int e    = t2 & 15;
    int lane = (t2 >> 4) & 31;
    int c    = (t2 >> 9) & 3;
    int t    = (t2 >> 11) & 7;
    int k = c * 32 + ((lane < 16) ? 0 : 8) + e + ((e < 8) ? 0 : 8);
    int n = t * 16 + (lane & 15);
    wfragO[t2] = (_Float16)out_w[k * 128 + n];
  }
}

// ---------------------------------------------------------------------------
// K1: h0 = ReLU(LayerNorm(features @ in_w + in_b))  -> f16
// ---------------------------------------------------------------------------
__global__ __launch_bounds__(128)
void inproj_kernel(const float* __restrict__ feat, const float* __restrict__ w,
                   const float* __restrict__ b, const float* __restrict__ g,
                   const float* __restrict__ bb, _Float16* __restrict__ h0) {
  __shared__ float s1[128];
  __shared__ float s2[128];
  int row = blockIdx.x, tid = threadIdx.x;
  const float* fr = feat + (long)row * INDIM;
  float x = b[tid];
#pragma unroll
  for (int k = 0; k < INDIM; ++k) x += fr[k] * w[k * 128 + tid];
  s1[tid] = x; s2[tid] = x * x;
  __syncthreads();
  for (int off = 64; off > 0; off >>= 1) {
    if (tid < off) { s1[tid] += s1[tid + off]; s2[tid] += s2[tid + off]; }
    __syncthreads();
  }
  float mean = s1[0] * (1.f / 128.f);
  float var  = s2[0] * (1.f / 128.f) - mean * mean;
  float rstd = rsqrtf(var + LN_EPS);
  float y = (x - mean) * rstd * g[tid] + bb[tid];
  h0[(long)row * DIM + tid] = (_Float16)fmaxf(y, 0.f);
}

// ---------------------------------------------------------------------------
// K2: one graph-conv layer.  One wave per group of MT row-tiles (64 rows).
// B fragments loaded once per (source, k-chunk) and reused across MT tiles:
// 4x less weight traffic, and 32 independent WMMAs per step for the scheduler.
// ---------------------------------------------------------------------------
__global__ __launch_bounds__(256)
void layer_kernel(const _Float16* __restrict__ hin, _Float16* __restrict__ hout,
                  const int* __restrict__ nexti, const int* __restrict__ previ,
                  const int* __restrict__ matei,
                  const _Float16* __restrict__ wfrag,   // this layer's fragments
                  const float* __restrict__ bsl,        // [4][128]
                  const float* __restrict__ lng, const float* __restrict__ lnb) {
  __shared__ float lds[8 * 16 * 132];
  int wave = threadIdx.x >> 5, lane = threadIdx.x & 31;
  int grp = blockIdx.x * 8 + wave;
  if (grp >= NGROUPS) return;
  int nc = lane & 15;
  bool lo_half = lane < 16;
  int tilebase = grp * MT;

  bool valid[MT];
  int idxv[4][MT];
#pragma unroll
  for (int mt = 0; mt < MT; ++mt) {
    int tile = tilebase + mt;
    valid[mt] = (tile < NTILES);
    int tc = valid[mt] ? tile : (NTILES - 1);
    int rg = tc * 16 + nc;
    idxv[0][mt] = rg;
    idxv[1][mt] = nexti[rg];
    idxv[2][mt] = previ[rg];
    idxv[3][mt] = matei[rg];
  }

  v8f acc[MT][8];
#pragma unroll
  for (int t = 0; t < 8; ++t) {
    int col = t * 16 + nc;
    float bsum = bsl[col] + bsl[128 + col] + bsl[256 + col] + bsl[384 + col];
#pragma unroll
    for (int mt = 0; mt < MT; ++mt)
#pragma unroll
      for (int r = 0; r < 8; ++r) acc[mt][t][r] = bsum;
  }

#pragma unroll
  for (int s = 0; s < 4; ++s) {
#pragma unroll
    for (int c = 0; c < 4; ++c) {
      const v16h* wb = reinterpret_cast<const v16h*>(wfrag) + s * 1024 + c * 32 + lane;
      v16h bfr[8];
#pragma unroll
      for (int t = 0; t < 8; ++t) bfr[t] = wb[t * 128];
      int kb = c * 32 + (lo_half ? 0 : 8);
#pragma unroll
      for (int mt = 0; mt < MT; ++mt) {
        const _Float16* rp = hin + (long)idxv[s][mt] * DIM;
        v8h alo = *reinterpret_cast<const v8h*>(rp + kb);
        v8h ahi = *reinterpret_cast<const v8h*>(rp + kb + 16);
        v16h a;
#pragma unroll
        for (int e = 0; e < 8; ++e) { a[e] = alo[e]; a[e + 8] = ahi[e]; }
#pragma unroll
        for (int t = 0; t < 8; ++t) {
          acc[mt][t] = __builtin_amdgcn_wmma_f32_16x16x32_f16(
              false, a, false, bfr[t], (short)0, acc[mt][t], false, false);
        }
      }
    }
  }

  // epilogue: per sub-tile, ReLU -> LDS stage -> LayerNorm -> residual -> f16
  float* my = lds + wave * (16 * 132);
  int c0 = lane * 4;
  float gv[4], bv[4];
#pragma unroll
  for (int j = 0; j < 4; ++j) { gv[j] = lng[c0 + j]; bv[j] = lnb[c0 + j]; }

#pragma unroll
  for (int mt = 0; mt < MT; ++mt) {
    if (!valid[mt]) continue;             // wave-uniform branch
    int base = (tilebase + mt) * 16;
#pragma unroll
    for (int t = 0; t < 8; ++t) {
      int col = t * 16 + nc;
#pragma unroll
      for (int r = 0; r < 8; ++r) {
        int m = lo_half ? r : r + 8;
        my[m * 132 + col] = fmaxf(acc[mt][t][r], 0.0f);
      }
    }
    // cross-lane LDS dependence within the wave: drain DS ops, fence scheduler
    asm volatile("s_wait_dscnt 0x0" ::: "memory");
    __builtin_amdgcn_wave_barrier();

    for (int m = 0; m < 16; ++m) {
      float xs[4];
#pragma unroll
      for (int j = 0; j < 4; ++j) xs[j] = my[m * 132 + c0 + j];
      float ls = xs[0] + xs[1] + xs[2] + xs[3];
      float lq = xs[0] * xs[0] + xs[1] * xs[1] + xs[2] * xs[2] + xs[3] * xs[3];
#pragma unroll
      for (int off = 16; off > 0; off >>= 1) {
        ls += __shfl_xor(ls, off, 32);
        lq += __shfl_xor(lq, off, 32);
      }
      float mean = ls * (1.f / 128.f);
      float var  = lq * (1.f / 128.f) - mean * mean;
      float rstd = rsqrtf(var + LN_EPS);
      long rg = base + m;
      v4h hold = *reinterpret_cast<const v4h*>(hin + rg * (long)DIM + c0);
      v4h ov;
#pragma unroll
      for (int j = 0; j < 4; ++j) {
        float y = (xs[j] - mean) * rstd * gv[j] + bv[j];
        ov[j] = (_Float16)(y + (float)hold[j]);
      }
      *reinterpret_cast<v4h*>(hout + rg * (long)DIM + c0) = ov;
    }
    asm volatile("" ::: "memory");        // keep next mt's stores below these loads
  }
}

// ---------------------------------------------------------------------------
// K3: coedge_emb = h @ out_w + out_b, scattered into per-face sums via atomics
// ---------------------------------------------------------------------------
__global__ __launch_bounds__(256)
void outproj_kernel(const _Float16* __restrict__ hin,
                    const _Float16* __restrict__ wfrag,
                    const float* __restrict__ outb,
                    const int* __restrict__ faceidx,
                    float* __restrict__ face_sum, float* __restrict__ counts) {
  int wave = threadIdx.x >> 5, lane = threadIdx.x & 31;
  int tile = blockIdx.x * 8 + wave;
  if (tile >= NTILES) return;
  int base = tile * 16;
  int nc = lane & 15;
  bool lo_half = lane < 16;
  int rowg = base + nc;

  v8f acc[8];
#pragma unroll
  for (int t = 0; t < 8; ++t) {
    float bvv = outb[t * 16 + nc];
#pragma unroll
    for (int r = 0; r < 8; ++r) acc[t][r] = bvv;
  }
  const _Float16* rp = hin + (long)rowg * DIM;
#pragma unroll
  for (int c = 0; c < 4; ++c) {
    int kb = c * 32 + (lo_half ? 0 : 8);
    v8h alo = *reinterpret_cast<const v8h*>(rp + kb);
    v8h ahi = *reinterpret_cast<const v8h*>(rp + kb + 16);
    v16h a;
#pragma unroll
    for (int e = 0; e < 8; ++e) { a[e] = alo[e]; a[e + 8] = ahi[e]; }
    const v16h* wb = reinterpret_cast<const v16h*>(wfrag) + c * 32 + lane;
#pragma unroll
    for (int t = 0; t < 8; ++t) {
      v16h bm = wb[t * 128];
      acc[t] = __builtin_amdgcn_wmma_f32_16x16x32_f16(
          false, a, false, bm, (short)0, acc[t], false, false);
    }
  }
#pragma unroll
  for (int r = 0; r < 8; ++r) {
    int m = lo_half ? r : r + 8;
    int rg = base + m;
    int f = faceidx[rg];
    long fb = (long)f * DIM;
#pragma unroll
    for (int t = 0; t < 8; ++t)
      atomicAdd(face_sum + fb + t * 16 + nc, acc[t][r]);
    if (nc == 0) atomicAdd(counts + f, 1.0f);
  }
}

// ---------------------------------------------------------------------------
// small epilogue kernels
// ---------------------------------------------------------------------------
__global__ void zero_kernel(float* __restrict__ p, int n) {
  int i = blockIdx.x * blockDim.x + threadIdx.x;
  if (i < n) p[i] = 0.f;
}

__global__ void faceemb_kernel(const float* __restrict__ fsum,
                               const float* __restrict__ counts,
                               float* __restrict__ out) {
  int i = blockIdx.x * blockDim.x + threadIdx.x;
  if (i < NFACE * DIM) {
    int f = i >> 7;
    out[i] = fsum[i] / fmaxf(counts[f], 1.0f);
  }
}

__global__ __launch_bounds__(128)
void gate_kernel(const float* __restrict__ femb, const float* __restrict__ g1w,
                 const float* __restrict__ g1b, const float* __restrict__ g2w,
                 const float* __restrict__ g2b, float* __restrict__ scores) {
  __shared__ float fe[128];
  __shared__ float red[128];
  int f = blockIdx.x, tid = threadIdx.x;
  fe[tid] = femb[(long)f * DIM + tid];
  __syncthreads();
  float s = g1b[tid];
#pragma unroll 4
  for (int k = 0; k < 128; ++k) s += fe[k] * g1w[k * 128 + tid];
  red[tid] = tanhf(s) * g2w[tid];
  __syncthreads();
  for (int off = 64; off > 0; off >>= 1) {
    if (tid < off) red[tid] += red[tid + off];
    __syncthreads();
  }
  if (tid == 0) scores[f] = red[0] + g2b[0];
}

__global__ __launch_bounds__(1024)
void softmax_kernel(const float* __restrict__ scores, float* __restrict__ attn) {
  __shared__ float red[1024];
  int tid = threadIdx.x;
  float m = -3.4e38f;
  for (int i = tid; i < NFACE; i += 1024) m = fmaxf(m, scores[i]);
  red[tid] = m; __syncthreads();
  for (int off = 512; off > 0; off >>= 1) {
    if (tid < off) red[tid] = fmaxf(red[tid], red[tid + off]);
    __syncthreads();
  }
  float M = red[0];
  __syncthreads();
  float s = 0.f;
  for (int i = tid; i < NFACE; i += 1024) s += expf(scores[i] - M);
  red[tid] = s; __syncthreads();
  for (int off = 512; off > 0; off >>= 1) {
    if (tid < off) red[tid] += red[tid + off];
    __syncthreads();
  }
  float inv = 1.f / red[0];
  for (int i = tid; i < NFACE; i += 1024) attn[i] = expf(scores[i] - M) * inv;
}

__global__ __launch_bounds__(128)
void graph_kernel(const float* __restrict__ attn, const float* __restrict__ femb,
                  float* __restrict__ gout) {
  int c = threadIdx.x;
  float acc = 0.f;
  for (int f = 0; f < NFACE; ++f) acc += attn[f] * femb[(long)f * DIM + c];
  gout[c] = acc;
}

// ---------------------------------------------------------------------------
extern "C" void kernel_launch(void* const* d_in, const int* in_sizes, int n_in,
                              void* d_out, int out_size, void* d_ws, size_t ws_size,
                              hipStream_t stream) {
  const float* features = (const float*)d_in[0];
  const int*   nexti    = (const int*)d_in[1];
  const int*   previ    = (const int*)d_in[2];
  const int*   matei    = (const int*)d_in[3];
  const int*   facei    = (const int*)d_in[4];
  const float* in_w     = (const float*)d_in[5];
  const float* in_b     = (const float*)d_in[6];
  const float* in_ln_g  = (const float*)d_in[7];
  const float* in_ln_b  = (const float*)d_in[8];
  const float* Ws       = (const float*)d_in[9];
  const float* bs       = (const float*)d_in[10];
  const float* ln_g     = (const float*)d_in[11];
  const float* ln_b     = (const float*)d_in[12];
  const float* out_w    = (const float*)d_in[13];
  const float* out_b    = (const float*)d_in[14];
  const float* g1_w     = (const float*)d_in[15];
  const float* g1_b     = (const float*)d_in[16];
  const float* g2_w     = (const float*)d_in[17];
  const float* g2_b     = (const float*)d_in[18];
  float* out = (float*)d_out;

  char* ws = (char*)d_ws;
  size_t off = 0;
  auto alloc = [&](size_t bytes) -> void* {
    void* p = ws + off;
    off = (off + bytes + 255) & ~(size_t)255;
    return p;
  };
  _Float16* wfragL   = (_Float16*)alloc((size_t)ALL_LAYER_HALVES * 2);
  _Float16* wfragO   = (_Float16*)alloc((size_t)OUT_FRAG_HALVES * 2);
  _Float16* hA       = (_Float16*)alloc((size_t)NCO * DIM * 2);
  _Float16* hB       = (_Float16*)alloc((size_t)NCO * DIM * 2);
  float*    face_sum = (float*)alloc((size_t)NFACE * DIM * 4);  // counts follows contiguously
  float*    counts   = (float*)alloc((size_t)NFACE * 4);
  float*    scores   = (float*)alloc((size_t)NFACE * 4);
  float*    attn     = (float*)alloc((size_t)NFACE * 4);
  (void)counts;

  // K0: pack weights into WMMA fragment order
  int packTotal = ALL_LAYER_HALVES + OUT_FRAG_HALVES;
  pack_kernel<<<(packTotal + 255) / 256, 256, 0, stream>>>(Ws, out_w, wfragL, wfragO);

  // K1: input projection + LN + ReLU
  inproj_kernel<<<NCO, 128, 0, stream>>>(features, in_w, in_b, in_ln_g, in_ln_b, hA);

  // K2 x6: graph conv layers (ping-pong f16 h buffers)
  _Float16* cur = hA;
  _Float16* nxt = hB;
  int gblocks = (NGROUPS + 7) / 8;
  for (int l = 0; l < NLAYERS; ++l) {
    layer_kernel<<<gblocks, 256, 0, stream>>>(
        cur, nxt, nexti, previ, matei,
        wfragL + (size_t)l * LAYER_FRAG_HALVES,
        bs + l * 4 * 128, ln_g + l * 128, ln_b + l * 128);
    _Float16* t = cur; cur = nxt; nxt = t;
  }

  // zero face accumulators (face_sum + counts are contiguous)
  int zN = NFACE * DIM + NFACE;
  zero_kernel<<<(zN + 255) / 256, 256, 0, stream>>>(face_sum, zN);

  // K3: output projection + per-face atomic accumulation
  int blocks = (NTILES + 7) / 8;
  outproj_kernel<<<blocks, 256, 0, stream>>>(cur, wfragO, out_b, facei, face_sum, counts);

  // K4: face mean pool -> d_out[0 : 50000*128]
  faceemb_kernel<<<(NFACE * DIM + 255) / 256, 256, 0, stream>>>(face_sum, counts, out);

  // K5: attention gate scores
  gate_kernel<<<NFACE, 128, 0, stream>>>(out, g1_w, g1_b, g2_w, g2_b, scores);

  // K6: softmax over faces
  softmax_kernel<<<1, 1024, 0, stream>>>(scores, attn);

  // K7: graph embedding -> d_out[50000*128 : +128]
  graph_kernel<<<1, 128, 0, stream>>>(attn, out, out + (size_t)NFACE * DIM);
}
